// ComplexGRUCell_19533511262337
// MI455X (gfx1250) — matrix operand order repaired and checked by reading, not compiled, not run on hardware
//
#include <hip/hip_runtime.h>
#include <math.h>

// ---------------------------------------------------------------------------
// ComplexGRUCell fused kernel for MI455X (gfx1250, wave32, WMMA).
//
// 24 GEMMs [B,256]x[256,256] (~206 GFLOP) + elementwise GRU epilogue.
// Compute-bound => f16 v_wmma_f32_16x16x32_f16 path, fp32 accumulate.
// Negated A-fragments are pre-staged in LDS to avoid WMMA->VALU hazard NOPs.
// ---------------------------------------------------------------------------

typedef __attribute__((ext_vector_type(16))) _Float16 v16h;
typedef __attribute__((ext_vector_type(8)))  _Float16 v8h;
typedef __attribute__((ext_vector_type(4)))  _Float16 v4h;
typedef __attribute__((ext_vector_type(8)))  float    v8f;
typedef unsigned int u32;

#define HDIM  256
#define PITCH 264   // LDS row pitch in halves: 256 + 8 pad -> conflict-free b128 reads

// ---------------------------------------------------------------------------

__device__ __forceinline__ v8f wmma_f16(v16h a, v16h b, v8f c) {
  // D = A(16x32 f16) * B(32x16 f16) + C(16x16 f32)
  return __builtin_amdgcn_wmma_f32_16x16x32_f16(
      /*neg_a=*/false, a, /*neg_b=*/false, b,
      /*c_mod=*/(short)0, c, /*reuse_a=*/false, /*reuse_b=*/false);
}

// A-fragment 16x32 f16 from LDS: lane = M (lanes 0-15 == 16-31),
// lane-half lh: K = lh*16 + j (j=0..15 contiguous), matching B.
__device__ __forceinline__ v16h load_a_frag(const _Float16* lds, int nl, int lh, int kt) {
  const _Float16* p0 = lds + nl * PITCH + kt * 32 + lh * 16;
  v8h a0 = *(const v8h*)p0;
  v8h a1 = *(const v8h*)(p0 + 8);
  return __builtin_shufflevector(a0, a1, 0,1,2,3,4,5,6,7,8,9,10,11,12,13,14,15);
}

// B-fragment 32x16 f16 from pre-converted weights (row-major [H=256][K=256] f16):
// lane = N column, K = lh*16 + j (j=0..15 contiguous) -> one 32B load per lane.
__device__ __forceinline__ v16h load_b_frag(const _Float16* __restrict__ W,
                                            int slot, int ncol, int lh, int kt) {
  const _Float16* p = W + (size_t)slot * (HDIM * HDIM)
                        + (size_t)ncol * HDIM + kt * 32 + lh * 16;
  return *(const v16h*)p;   // 32B aligned by construction -> 2x global_load_b128
}

__device__ __forceinline__ float sigmoidf_(float x) {
  return 1.0f / (1.0f + __expf(-x));   // v_exp_f32 trans path
}

__device__ __forceinline__ float fast_tanh(float x) {
#if __has_builtin(__builtin_amdgcn_tanhf)
  return __builtin_amdgcn_tanhf(x);        // v_tanh_f32 (gfx1250 TRANS op)
#elif __has_builtin(__builtin_amdgcn_tanh_f32)
  return __builtin_amdgcn_tanh_f32(x);
#else
  return tanhf(x);
#endif
}

// stage 8 consecutive fp32 -> 8 f16 into LDS
__device__ __forceinline__ void stage8(const float* __restrict__ s, _Float16* d) {
  float4 a0 = *(const float4*)s;
  float4 a1 = *(const float4*)(s + 4);
  v8h c = { (_Float16)a0.x, (_Float16)a0.y, (_Float16)a0.z, (_Float16)a0.w,
            (_Float16)a1.x, (_Float16)a1.y, (_Float16)a1.z, (_Float16)a1.w };
  *(v8h*)d = c;
}

// stage 8 fp32 -> f16 into LDS twice: positive copy and negated copy
__device__ __forceinline__ void stage8pn(const float* __restrict__ s,
                                         _Float16* dp, _Float16* dn) {
  float4 a0 = *(const float4*)s;
  float4 a1 = *(const float4*)(s + 4);
  v8h c = { (_Float16)a0.x, (_Float16)a0.y, (_Float16)a0.z, (_Float16)a0.w,
            (_Float16)a1.x, (_Float16)a1.y, (_Float16)a1.z, (_Float16)a1.w };
  v8h n = { (_Float16)-a0.x, (_Float16)-a0.y, (_Float16)-a0.z, (_Float16)-a0.w,
            (_Float16)-a1.x, (_Float16)-a1.y, (_Float16)-a1.z, (_Float16)-a1.w };
  *(v8h*)dp = c;
  *(v8h*)dn = n;
}

// ---------------------------------------------------------------------------
// prep: fp32 [256x256] weight matrix -> f16 (same row-major layout)
// ---------------------------------------------------------------------------
__global__ void cvt_w16(const float* __restrict__ s, _Float16* __restrict__ d) {
  const int i = (blockIdx.x * blockDim.x + threadIdx.x) * 4;
  float4 v = *(const float4*)(s + i);
  v4h o = { (_Float16)v.x, (_Float16)v.y, (_Float16)v.z, (_Float16)v.w };
  *(v4h*)(d + i) = o;
}

// ---------------------------------------------------------------------------
// main fused kernel: 1 WG = 16 batch rows x all 256 output cols.
// 16 waves; wave w owns N-tile [w*16, w*16+16).
// 12 accumulators per wave: for gate g: {Xreal, Ximag, Hreal, Himag}.
// Weight slot order in d_ws: c*2+{Wr,Wi}, c in (w1,r1,w2,r2,w3,r3).
// ---------------------------------------------------------------------------
__global__ __launch_bounds__(512) void gru_main(
    const float* __restrict__ xr, const float* __restrict__ xi,
    const float* __restrict__ hr, const float* __restrict__ hi,
    const _Float16* __restrict__ W,
    const float* __restrict__ w1br, const float* __restrict__ w1bi,
    const float* __restrict__ r1br, const float* __restrict__ r1bi,
    const float* __restrict__ w2br, const float* __restrict__ w2bi,
    const float* __restrict__ r2br, const float* __restrict__ r2bi,
    const float* __restrict__ w3br, const float* __restrict__ w3bi,
    const float* __restrict__ r3br, const float* __restrict__ r3bi,
    float* __restrict__ out, int B)
{
  __shared__ _Float16 lxr [16 * PITCH];
  __shared__ _Float16 lxi [16 * PITCH];
  __shared__ _Float16 lnxi[16 * PITCH];   // -xi (f16)
  __shared__ _Float16 lhr [16 * PITCH];
  __shared__ _Float16 lhi [16 * PITCH];
  __shared__ _Float16 lnhi[16 * PITCH];   // -hi (f16)

  const int tid   = threadIdx.x;
  const int mbase = blockIdx.x * 16;

  // ---- stage activations fp32 -> f16 LDS (each thread: 8 elems x 4 matrices,
  //      plus negated copies of xi/hi)
  {
    const int e   = tid * 8;          // 512 threads * 8 = 16*256
    const int row = e >> 8;
    const int col = e & 255;
    const size_t g = (size_t)(mbase + row) * HDIM + col;
    const int    l = row * PITCH + col;
    stage8  (xr + g, lxr + l);
    stage8pn(xi + g, lxi + l, lnxi + l);
    stage8  (hr + g, lhr + l);
    stage8pn(hi + g, lhi + l, lnhi + l);
  }
  __syncthreads();

  const int lane = tid & 31;
  const int wave = tid >> 5;          // N-tile id 0..15
  const int nl   = lane & 15;
  const int lh   = lane >> 4;
  const int ncol = wave * 16 + nl;    // this lane's output column (C/D lane = N)

  v8f acc[12];
#pragma unroll
  for (int i = 0; i < 12; ++i)
    acc[i] = (v8f){0.f, 0.f, 0.f, 0.f, 0.f, 0.f, 0.f, 0.f};

  // ---- K loop: 8 steps of K=32, 24 WMMAs per step
#pragma unroll 2
  for (int kt = 0; kt < 8; ++kt) {
    v16h axr = load_a_frag(lxr,  nl, lh, kt);
    v16h axi = load_a_frag(lxi,  nl, lh, kt);
    v16h nxi = load_a_frag(lnxi, nl, lh, kt);
    v16h ahr = load_a_frag(lhr,  nl, lh, kt);
    v16h ahi = load_a_frag(lhi,  nl, lh, kt);
    v16h nhi = load_a_frag(lnhi, nl, lh, kt);
#pragma unroll
    for (int g = 0; g < 3; ++g) {
      v16h xWr = load_b_frag(W, 4 * g + 0, ncol, lh, kt);
      v16h xWi = load_b_frag(W, 4 * g + 1, ncol, lh, kt);
      v16h hWr = load_b_frag(W, 4 * g + 2, ncol, lh, kt);
      v16h hWi = load_b_frag(W, 4 * g + 3, ncol, lh, kt);
      // real(x) = xr@Wr - xi@Wi ; imag(x) = xi@Wr + xr@Wi   (same for h)
      acc[4*g+0] = wmma_f16(axr, xWr, acc[4*g+0]);
      acc[4*g+0] = wmma_f16(nxi, xWi, acc[4*g+0]);
      acc[4*g+1] = wmma_f16(axi, xWr, acc[4*g+1]);
      acc[4*g+1] = wmma_f16(axr, xWi, acc[4*g+1]);
      acc[4*g+2] = wmma_f16(ahr, hWr, acc[4*g+2]);
      acc[4*g+2] = wmma_f16(nhi, hWi, acc[4*g+2]);
      acc[4*g+3] = wmma_f16(ahi, hWr, acc[4*g+3]);
      acc[4*g+3] = wmma_f16(ahr, hWi, acc[4*g+3]);
    }
  }

  // ---- epilogue: each lane owns exactly one h column (ncol)
  const float b1r  = (w1br[ncol] - w1bi[ncol]) + (r1br[ncol] - r1bi[ncol]);
  const float b1i  = (w1br[ncol] + w1bi[ncol]) + (r1br[ncol] + r1bi[ncol]);
  const float b2r  = (w2br[ncol] - w2bi[ncol]) + (r2br[ncol] - r2bi[ncol]);
  const float b2i  = (w2br[ncol] + w2bi[ncol]) + (r2br[ncol] + r2bi[ncol]);
  const float bx3r = w3br[ncol] - w3bi[ncol];
  const float bx3i = w3br[ncol] + w3bi[ncol];
  const float bh3r = r3br[ncol] - r3bi[ncol];
  const float bh3i = r3br[ncol] + r3bi[ncol];

  const size_t imag_off = (size_t)B * HDIM;
#pragma unroll
  for (int r = 0; r < 8; ++r) {
    const int m = r + 8 * lh;       // C/D layout: VGPR r -> M = r + 8*lane_half
    float Rr = sigmoidf_(acc[0][r] + acc[2][r] + b1r);
    float Ri = sigmoidf_(acc[1][r] + acc[3][r] + b1i);
    float Zr = sigmoidf_(acc[4][r] + acc[6][r] + b2r);
    float Zi = sigmoidf_(acc[5][r] + acc[7][r] + b2i);
    float g3r = acc[10][r] + bh3r;
    float g3i = acc[11][r] + bh3i;
    float h3r = Rr * g3r - Ri * g3i;          // r * (h@r3)
    float h3i = Rr * g3i + Ri * g3r;
    float Nr  = fast_tanh(acc[8][r] + bx3r + h3r);
    float Ni  = fast_tanh(acc[9][r] + bx3i + h3i);
    float hvr = (float)lhr[m * PITCH + ncol]; // original h (f16 copy)
    float hvi = (float)lhi[m * PITCH + ncol];
    // (1 - z~) * n + z * h, complex
    float outr = (1.0f - Zr) * Nr + Zi * Ni + Zr * hvr - Zi * hvi;
    float outi = (1.0f - Zr) * Ni - Zi * Nr + Zr * hvi + Zi * hvr;
    const size_t o = (size_t)(mbase + m) * HDIM + ncol;
    out[o]            = outr;
    out[imag_off + o] = outi;
  }
}

// ---------------------------------------------------------------------------

extern "C" void kernel_launch(void* const* d_in, const int* in_sizes, int n_in,
                              void* d_out, int out_size, void* d_ws, size_t ws_size,
                              hipStream_t stream) {
  (void)n_in; (void)out_size; (void)ws_size;

  const float* xr = (const float*)d_in[0];
  const float* xi = (const float*)d_in[1];
  const float* hr = (const float*)d_in[2];
  const float* hi = (const float*)d_in[3];
  const int B = in_sizes[0] / HDIM;   // 65536

  // d_ws: 12 f16 weight matrices, slot c*2+{Wr=0,Wi=1}, c = w1,r1,w2,r2,w3,r3
  _Float16* W16 = (_Float16*)d_ws;    // needs 12*256*256*2 = 1.5 MB
  for (int c = 0; c < 6; ++c) {
    cvt_w16<<<64, 256, 0, stream>>>((const float*)d_in[4 + 4 * c],
                                    W16 + (size_t)(2 * c) * (HDIM * HDIM));
    cvt_w16<<<64, 256, 0, stream>>>((const float*)d_in[5 + 4 * c],
                                    W16 + (size_t)(2 * c + 1) * (HDIM * HDIM));
  }

  gru_main<<<B / 16, 512, 0, stream>>>(
      xr, xi, hr, hi, W16,
      (const float*)d_in[6],  (const float*)d_in[7],    // w1br, w1bi
      (const float*)d_in[10], (const float*)d_in[11],   // r1br, r1bi
      (const float*)d_in[14], (const float*)d_in[15],   // w2br, w2bi
      (const float*)d_in[18], (const float*)d_in[19],   // r2br, r2bi
      (const float*)d_in[22], (const float*)d_in[23],   // w3br, w3bi
      (const float*)d_in[26], (const float*)d_in[27],   // r3br, r3bi
      (float*)d_out, B);
}
